// TFAttention_2989297238416
// MI455X (gfx1250) — matrix operand order repaired and checked by reading, not compiled, not run on hardware
//
#include <hip/hip_runtime.h>
#include <hip/hip_bf16.h>

// ---------------------------------------------------------------------------
// TF-attention for MI455X (gfx1250): all GEMMs via V_WMMA_F32_16X16X32_F16.
// B=4, C=64, T=1024, F=128, H=4, HID=4, CH=16 -> N=16 batches, DQ=512, DV=2048.
// All WMMA fragments are gathered with 128-bit LDS loads (ds_load_b128):
// A tiles staged row-major [M][K]; B tiles staged transposed [N][K].
// ---------------------------------------------------------------------------

typedef __attribute__((ext_vector_type(16))) _Float16 v16h;
typedef __attribute__((ext_vector_type(8)))  _Float16 v8h;
typedef __attribute__((ext_vector_type(4)))  _Float16 v4h;
typedef __attribute__((ext_vector_type(8)))  float    v8f;
typedef __attribute__((ext_vector_type(4)))  float    v4f;

#define DEV_INLINE __device__ __forceinline__

DEV_INLINE v8f wmma_f16(v16h a, v16h b, v8f c) {
  return __builtin_amdgcn_wmma_f32_16x16x32_f16(
      false, a, false, b, (short)0, c, false, false);
}

DEV_INLINE v16h join16(v8h lo, v8h hi) {
  return __builtin_shufflevector(lo, hi, 0, 1, 2, 3, 4, 5, 6, 7,
                                 8, 9, 10, 11, 12, 13, 14, 15);
}

// A fragment (16x32 f16) from row-major [M][K], ld multiple of 8.
// lane l: M=l&15; K halves hi*8..hi*8+7 and 16+hi*8..  -> two b128 loads.
DEV_INLINE v16h load_a_frag(const _Float16* src, int ld, int m0, int k0) {
  int l = threadIdx.x & 31, m = l & 15, hi = l >> 4;
  const _Float16* p = src + (size_t)(m0 + m) * ld + k0 + hi * 8;
  return join16(*(const v8h*)p, *(const v8h*)(p + 16));
}

// B fragment (32x16 f16) from TRANSPOSED [N][K] storage, ld multiple of 8.
// lane l: N=l&15; K halves hi*8..hi*8+7 and 16+hi*8..  -> two b128 loads.
DEV_INLINE v16h load_b_frag_nk(const _Float16* src, int ld, int n0, int k0) {
  int l = threadIdx.x & 31, n = l & 15, hi = l >> 4;
  const _Float16* p = src + (size_t)(n0 + n) * ld + k0 + hi * 8;
  return join16(*(const v8h*)p, *(const v8h*)(p + 16));
}

// Copy a 128x32 f16 tile (rows srcStride apart, cols k0..k0+31) into LDS
// with 128-bit transfers. 256 threads.
DEV_INLINE void copy_tile(_Float16* __restrict__ dst,
                          const _Float16* __restrict__ src,
                          int srcStride, int k0) {
  for (int e = threadIdx.x; e < 512; e += 256) {
    int r = e >> 2, c8 = (e & 3) * 8;
    *(v8h*)(dst + r * 32 + c8) =
        *(const v8h*)(src + (size_t)r * srcStride + k0 + c8);
    __builtin_prefetch(src + (size_t)r * srcStride + k0 + c8 + 32, 0, 1);
  }
}

// ---- block reductions (256 threads = 8 waves of 32) -----------------------

DEV_INLINE float block_reduce_sum(float v, float* scratch) {
#pragma unroll
  for (int off = 16; off > 0; off >>= 1) v += __shfl_down(v, off, 32);
  int wid = threadIdx.x >> 5;
  __syncthreads();
  if ((threadIdx.x & 31) == 0) scratch[wid] = v;
  __syncthreads();
  float total = 0.f;
#pragma unroll
  for (int i = 0; i < 8; ++i) total += scratch[i];
  return total;
}

DEV_INLINE float block_reduce_max(float v, float* scratch) {
#pragma unroll
  for (int off = 16; off > 0; off >>= 1) v = fmaxf(v, __shfl_down(v, off, 32));
  int wid = threadIdx.x >> 5;
  __syncthreads();
  if ((threadIdx.x & 31) == 0) scratch[wid] = v;
  __syncthreads();
  float total = -3.0e38f;
#pragma unroll
  for (int i = 0; i < 8; ++i) total = fmaxf(total, scratch[i]);
  return total;
}

// ---------------------------------------------------------------------------
// Kernel 1: fused QKV head-branches.  One block per (b,t).
// Stacked weights rows: [0..15]=Wq, [16..31]=Wk, [32..95]=Wv.
// GEMM 96x128x64 + bias + PReLU + 12 group ChannelNorms + gamma/beta.
// ---------------------------------------------------------------------------
__global__ void __launch_bounds__(256) qkv_kernel(
    const float* __restrict__ x,
    const float* __restrict__ Wq, const float* __restrict__ bq,
    const float* __restrict__ aq, const float* __restrict__ gq,
    const float* __restrict__ beq,
    const float* __restrict__ Wk, const float* __restrict__ bk,
    const float* __restrict__ ak, const float* __restrict__ gk,
    const float* __restrict__ bek,
    const float* __restrict__ Wv, const float* __restrict__ bv,
    const float* __restrict__ av, const float* __restrict__ gv,
    const float* __restrict__ bev,
    _Float16* __restrict__ qh, _Float16* __restrict__ kh,
    _Float16* __restrict__ vh) {
  const int T = 1024, F = 128;
  int blk = blockIdx.x;
  int b   = blk >> 10;
  int t   = blk & 1023;

  __shared__ alignas(16) union {
    struct { _Float16 Xt[128 * 64]; _Float16 W[96 * 64]; } in;  // 16+12 KB
    float y[96 * 128];                                          // 48 KB
  } sm;
  __shared__ float red[8];
  __shared__ float stats[24];

  // Stage Xᵀ[f][c] (B operand, transposed) from x[b][c][t][f].
  const float* xbt = x + ((size_t)b * 64 * T + t) * F;
  for (int ch = threadIdx.x; ch < 2048; ch += 256) {
    int c = ch >> 5, f0 = (ch & 31) * 4;
    v4f xv = *(const v4f*)(xbt + (size_t)c * T * F + f0);
#pragma unroll
    for (int j = 0; j < 4; ++j)
      sm.in.Xt[(f0 + j) * 64 + c] = (_Float16)xv[j];
  }
  // Stage stacked weights row-major [r][c] (A operand).
  for (int ch = threadIdx.x; ch < 1536; ch += 256) {
    int r = ch >> 4, c0 = (ch & 15) * 4;
    const float* wsrc = (r < 16) ? &Wq[r * 64]
                      : (r < 32) ? &Wk[(r - 16) * 64]
                                 : &Wv[(r - 32) * 64];
    v4f wv = *(const v4f*)(wsrc + c0);
    v4h wh;
#pragma unroll
    for (int j = 0; j < 4; ++j) wh[j] = (_Float16)wv[j];
    *(v4h*)(&sm.in.W[r * 64 + c0]) = wh;
  }
  __syncthreads();

  // 48 tiles (6 M x 8 N), 8 waves -> 6 tiles each; K = 64 (2 steps).
  int wid = threadIdx.x >> 5;
  v8f acc[6] = {};
#pragma unroll
  for (int j = 0; j < 6; ++j) {
    int tI = wid * 6 + j;
    int mt = tI >> 3, nt = tI & 7;
#pragma unroll
    for (int ks = 0; ks < 2; ++ks) {
      v16h a  = load_a_frag(sm.in.W, 64, mt * 16, ks * 32);
      v16h bb = load_b_frag_nk(sm.in.Xt, 64, nt * 16, ks * 32);
      acc[j]  = wmma_f16(a, bb, acc[j]);
    }
  }
  __syncthreads();  // Xt/W dead; sm.y overlaps.

  // bias + PReLU into LDS y[96][128].
  int l = threadIdx.x & 31, nn = l & 15, hi = l >> 4;
#pragma unroll
  for (int j = 0; j < 6; ++j) {
    int tI = wid * 6 + j;
    int mt = tI >> 3, nt = tI & 7;
#pragma unroll
    for (int r = 0; r < 8; ++r) {
      int row = mt * 16 + r + 8 * hi;
      int col = nt * 16 + nn;
      float bias, alpha;
      if (row < 16)      { bias = bq[row];      alpha = aq[row >> 2]; }
      else if (row < 32) { bias = bk[row - 16]; alpha = ak[(row - 16) >> 2]; }
      else               { bias = bv[row - 32]; alpha = av[(row - 32) >> 4]; }
      float v = acc[j][r] + bias;
      v = v > 0.f ? v : alpha * v;
      sm.y[row * 128 + col] = v;
    }
  }
  __syncthreads();

  // 12 group norms over contiguous row ranges.
  for (int g = 0; g < 12; ++g) {
    int row0, nrows;
    if (g < 4)      { row0 = g * 4;             nrows = 4;  }
    else if (g < 8) { row0 = 16 + (g - 4) * 4;  nrows = 4;  }
    else            { row0 = 32 + (g - 8) * 16; nrows = 16; }
    int cnt = nrows * 128;
    float s = 0.f, ss = 0.f;
    for (int e = threadIdx.x; e < cnt; e += 256) {
      float v = sm.y[row0 * 128 + e];
      s += v; ss += v * v;
    }
    s  = block_reduce_sum(s, red);
    ss = block_reduce_sum(ss, red);
    if (threadIdx.x == 0) {
      float mean = s / (float)cnt;
      float var  = fmaxf(ss / (float)cnt - mean * mean, 0.f);
      stats[g * 2]     = mean;
      stats[g * 2 + 1] = rsqrtf(var + 1e-5f);
    }
  }
  __syncthreads();

  // gamma/beta + vectorized f16 stores into attention layouts.
  for (int cidx = threadIdx.x; cidx < 1536; cidx += 256) {
    int row = cidx >> 4;
    int f0  = (cidx & 15) * 8;
    const float* yrow = &sm.y[row * 128 + f0];
    v8h outv;
    if (row < 16) {
      int h = row >> 2, o = row & 3, g = h;
      float mean = stats[g * 2], rstd = stats[g * 2 + 1];
      const float* gg = &gq[(h * 4 + o) * 128 + f0];
      const float* bb = &beq[(h * 4 + o) * 128 + f0];
#pragma unroll
      for (int j = 0; j < 8; ++j)
        outv[j] = (_Float16)((yrow[j] - mean) * rstd * gg[j] + bb[j]);
      int n = h * 4 + b;
      *(v8h*)(&qh[((size_t)n * 1024 + t) * 512 + o * 128 + f0]) = outv;
    } else if (row < 32) {
      int rr = row - 16, h = rr >> 2, o = rr & 3, g = 4 + h;
      float mean = stats[g * 2], rstd = stats[g * 2 + 1];
      const float* gg = &gk[(h * 4 + o) * 128 + f0];
      const float* bb = &bek[(h * 4 + o) * 128 + f0];
#pragma unroll
      for (int j = 0; j < 8; ++j)
        outv[j] = (_Float16)((yrow[j] - mean) * rstd * gg[j] + bb[j]);
      int n = h * 4 + b;
      *(v8h*)(&kh[((size_t)n * 1024 + t) * 512 + o * 128 + f0]) = outv;
    } else {
      int rr = row - 32, h = rr >> 4, chn = rr & 15, g = 8 + h;
      float mean = stats[g * 2], rstd = stats[g * 2 + 1];
      const float* gg = &gv[(h * 16 + chn) * 128 + f0];
      const float* bb = &bev[(h * 16 + chn) * 128 + f0];
#pragma unroll
      for (int j = 0; j < 8; ++j)
        outv[j] = (_Float16)((yrow[j] - mean) * rstd * gg[j] + bb[j]);
      int n = h * 4 + b;
      *(v8h*)(&vh[((size_t)n * 1024 + t) * 2048 + chn * 128 + f0]) = outv;
    }
  }
}

// ---------------------------------------------------------------------------
// Kernel 2: S = scale * Q Kᵀ, 128x128 tiles, K = 512, double-buffered LDS.
// grid = (8 col-tiles, 8 row-tiles, 16 batches), 256 threads.
// ---------------------------------------------------------------------------
__global__ void __launch_bounds__(256) scores_kernel(
    const _Float16* __restrict__ qh, const _Float16* __restrict__ kh,
    _Float16* __restrict__ S) {
  int n  = blockIdx.z;
  int i0 = blockIdx.y * 128;
  int j0 = blockIdx.x * 128;

  __shared__ alignas(16) _Float16 Qs[2][128 * 32];
  __shared__ alignas(16) _Float16 Ks[2][128 * 32];

  int wid = threadIdx.x >> 5;
  v8f acc[8] = {};

  const _Float16* qbase = qh + ((size_t)n * 1024 + i0) * 512;
  const _Float16* kbase = kh + ((size_t)n * 1024 + j0) * 512;

  copy_tile(Qs[0], qbase, 512, 0);
  copy_tile(Ks[0], kbase, 512, 0);

  for (int ks = 0; ks < 16; ++ks) {
    __syncthreads();
    int cur = ks & 1;
    if (ks + 1 < 16) {
      copy_tile(Qs[cur ^ 1], qbase, 512, (ks + 1) * 32);
      copy_tile(Ks[cur ^ 1], kbase, 512, (ks + 1) * 32);
    }
    v16h a = load_a_frag(Qs[cur], 32, wid * 16, 0);
#pragma unroll
    for (int nt = 0; nt < 8; ++nt) {
      v16h b  = load_b_frag_nk(Ks[cur], 32, nt * 16, 0);
      acc[nt] = wmma_f16(a, b, acc[nt]);
    }
  }

  const float scale = 0.04419417382415922f;  // 1/sqrt(512)
  int l = threadIdx.x & 31, nn = l & 15, hi = l >> 4;
#pragma unroll
  for (int nt = 0; nt < 8; ++nt)
#pragma unroll
    for (int r = 0; r < 8; ++r) {
      int row = i0 + wid * 16 + r + 8 * hi;
      int col = j0 + nt * 16 + nn;
      S[((size_t)n * 1024 + row) * 1024 + col] = (_Float16)(acc[nt][r] * scale);
    }
}

// ---------------------------------------------------------------------------
// Kernel 3: row softmax, in place on f16 S.  One block per row.
// ---------------------------------------------------------------------------
__global__ void __launch_bounds__(256) softmax_kernel(_Float16* __restrict__ S) {
  int row = blockIdx.x;
  int n   = blockIdx.y;
  _Float16* p = S + ((size_t)n * 1024 + row) * 1024 + threadIdx.x * 4;
  __shared__ float red[8];

  v4h vin = *(const v4h*)p;
  float v[4];
  float mx = -3.0e38f;
#pragma unroll
  for (int i = 0; i < 4; ++i) {
    v[i] = (float)vin[i];
    mx   = fmaxf(mx, v[i]);
  }
  mx = block_reduce_max(mx, red);
  float s = 0.f;
#pragma unroll
  for (int i = 0; i < 4; ++i) {
    v[i] = __expf(v[i] - mx);
    s += v[i];
  }
  s = block_reduce_sum(s, red);
  float inv = 1.f / s;
  v4h vout;
#pragma unroll
  for (int i = 0; i < 4; ++i) vout[i] = (_Float16)(v[i] * inv);
  *(v4h*)p = vout;
}

// ---------------------------------------------------------------------------
// Kernel 4: O = P V, 128x128 tiles, K = 1024, double-buffered LDS.
// V tile staged transposed [d][k] so B-fragments are contiguous.
// grid = (16 d-tiles, 8 t-tiles, 16 batches).
// ---------------------------------------------------------------------------
__global__ void __launch_bounds__(256) pv_kernel(
    const _Float16* __restrict__ P, const _Float16* __restrict__ vh,
    _Float16* __restrict__ oh) {
  int n  = blockIdx.z;
  int i0 = blockIdx.y * 128;
  int d0 = blockIdx.x * 128;

  __shared__ alignas(16) _Float16 Ps[2][128 * 32];
  __shared__ alignas(16) _Float16 Vt[2][128 * 32];

  int wid = threadIdx.x >> 5;
  v8f acc[8] = {};

  const _Float16* pbase = P + ((size_t)n * 1024 + i0) * 1024;
  const _Float16* vbase = vh + (size_t)n * 1024 * 2048 + d0;

  // Stage V slab transposed: Vt[cc][kk] = V[ks*32+kk][d0+cc].
  auto stage_v = [&](int buf, int ks) {
    for (int e = threadIdx.x; e < 512; e += 256) {
      int kk = e >> 4, c8 = (e & 15) * 8;
      v8h vv = *(const v8h*)(vbase + (size_t)(ks * 32 + kk) * 2048 + c8);
#pragma unroll
      for (int j = 0; j < 8; ++j) Vt[buf][(c8 + j) * 32 + kk] = vv[j];
    }
  };

  copy_tile(Ps[0], pbase, 1024, 0);
  stage_v(0, 0);

  for (int ks = 0; ks < 32; ++ks) {
    __syncthreads();
    int cur = ks & 1;
    if (ks + 1 < 32) {
      copy_tile(Ps[cur ^ 1], pbase, 1024, (ks + 1) * 32);
      stage_v(cur ^ 1, ks + 1);
    }
    v16h a = load_a_frag(Ps[cur], 32, wid * 16, 0);
#pragma unroll
    for (int nt = 0; nt < 8; ++nt) {
      v16h b  = load_b_frag_nk(Vt[cur], 32, nt * 16, 0);
      acc[nt] = wmma_f16(a, b, acc[nt]);
    }
  }

  int l = threadIdx.x & 31, nn = l & 15, hi = l >> 4;
  int h = n >> 2, b = n & 3;
#pragma unroll
  for (int nt = 0; nt < 8; ++nt)
#pragma unroll
    for (int r = 0; r < 8; ++r) {
      int t  = i0 + wid * 16 + r + 8 * hi;
      int d  = d0 + nt * 16 + nn;
      int chn = d >> 7, f = d & 127;
      int c  = h * 16 + chn;
      oh[(((size_t)b * 64 + c) * 1024 + t) * 128 + f] = (_Float16)acc[nt][r];
    }
}

// ---------------------------------------------------------------------------
// Kernel 5: projection 64x64 GEMM + PReLU + ChannelNorm(C,F) + residual.
// One block per (b,t).
// ---------------------------------------------------------------------------
__global__ void __launch_bounds__(256) proj_kernel(
    const _Float16* __restrict__ oh, const float* __restrict__ x,
    const float* __restrict__ Wp, const float* __restrict__ bp,
    const float* __restrict__ ap, const float* __restrict__ gp,
    const float* __restrict__ bep, float* __restrict__ out) {
  int blk = blockIdx.x;
  int b   = blk >> 10;
  int t   = blk & 1023;

  __shared__ alignas(16) union {
    struct { _Float16 Xt[128 * 64]; _Float16 W[64 * 64]; } in;  // 16+8 KB
    float y[64 * 128];                                          // 32 KB
  } sm;
  __shared__ float red[8];
  __shared__ float stats[2];

  // Stage Xᵀ[f][c] from oh[b][c][t][f] (128-bit reads, transposed stores).
  const _Float16* obt = oh + ((size_t)b * 64 * 1024 + t) * 128;
  for (int ch = threadIdx.x; ch < 1024; ch += 256) {
    int c = ch >> 4, f0 = (ch & 15) * 8;
    v8h ov = *(const v8h*)(obt + (size_t)c * 1024 * 128 + f0);
#pragma unroll
    for (int j = 0; j < 8; ++j) sm.in.Xt[(f0 + j) * 64 + c] = ov[j];
  }
  for (int ch = threadIdx.x; ch < 1024; ch += 256) {
    int r = ch >> 4, c0 = (ch & 15) * 4;
    v4f wv = *(const v4f*)(Wp + r * 64 + c0);
    v4h wh;
#pragma unroll
    for (int j = 0; j < 4; ++j) wh[j] = (_Float16)wv[j];
    *(v4h*)(&sm.in.W[r * 64 + c0]) = wh;
  }
  __syncthreads();

  // 32 tiles (4 M x 8 N), 8 waves -> 4 each; K = 64 (2 steps).
  int wid = threadIdx.x >> 5;
  v8f acc[4] = {};
#pragma unroll
  for (int j = 0; j < 4; ++j) {
    int tI = wid * 4 + j;
    int mt = tI >> 3, nt = tI & 7;
#pragma unroll
    for (int ks = 0; ks < 2; ++ks) {
      v16h a  = load_a_frag(sm.in.W, 64, mt * 16, ks * 32);
      v16h bb = load_b_frag_nk(sm.in.Xt, 64, nt * 16, ks * 32);
      acc[j]  = wmma_f16(a, bb, acc[j]);
    }
  }
  __syncthreads();

  int l = threadIdx.x & 31, nn = l & 15, hi = l >> 4;
  float alpha = ap[0];
#pragma unroll
  for (int j = 0; j < 4; ++j) {
    int tI = wid * 4 + j;
    int mt = tI >> 3, nt = tI & 7;
#pragma unroll
    for (int r = 0; r < 8; ++r) {
      int row = mt * 16 + r + 8 * hi;
      int col = nt * 16 + nn;
      float v = acc[j][r] + bp[row];
      v = v > 0.f ? v : alpha * v;
      sm.y[row * 128 + col] = v;
    }
  }
  __syncthreads();

  float s = 0.f, ss = 0.f;
  for (int e = threadIdx.x; e < 8192; e += 256) {
    float v = sm.y[e];
    s += v; ss += v * v;
  }
  s  = block_reduce_sum(s, red);
  ss = block_reduce_sum(ss, red);
  if (threadIdx.x == 0) {
    float mean = s / 8192.f;
    float var  = fmaxf(ss / 8192.f - mean * mean, 0.f);
    stats[0] = mean;
    stats[1] = rsqrtf(var + 1e-5f);
  }
  __syncthreads();

  float mean = stats[0], rstd = stats[1];
  const float* xbt   = x + ((size_t)b * 64 * 1024 + t) * 128;
  float*       obase = out + ((size_t)b * 64 * 1024 + t) * 128;
  for (int e4 = threadIdx.x; e4 < 2048; e4 += 256) {
    int c = e4 >> 5, f0 = (e4 & 31) * 4;
    int e = c * 128 + f0;
    v4f yv = *(const v4f*)(&sm.y[e]);
    v4f gg = *(const v4f*)(&gp[e]);
    v4f bb = *(const v4f*)(&bep[e]);
    v4f xv = *(const v4f*)(xbt + (size_t)c * 1024 * 128 + f0);
    v4f ov;
#pragma unroll
    for (int j = 0; j < 4; ++j)
      ov[j] = (yv[j] - mean) * rstd * gg[j] + bb[j] + xv[j];
    *(v4f*)(obase + (size_t)c * 1024 * 128 + f0) = ov;
  }
}

// ---------------------------------------------------------------------------
// Launch.  Workspace (f16): Q 16MB | K 16MB | V 64MB | S 32MB | O 64MB.
// ---------------------------------------------------------------------------
extern "C" void kernel_launch(void* const* d_in, const int* in_sizes, int n_in,
                              void* d_out, int out_size, void* d_ws,
                              size_t ws_size, hipStream_t stream) {
  (void)in_sizes; (void)n_in; (void)out_size; (void)ws_size;
  const float* x   = (const float*)d_in[0];
  const float* Wq  = (const float*)d_in[1];
  const float* bq  = (const float*)d_in[2];
  const float* aq  = (const float*)d_in[3];
  const float* gq  = (const float*)d_in[4];
  const float* beq = (const float*)d_in[5];
  const float* Wk  = (const float*)d_in[6];
  const float* bk  = (const float*)d_in[7];
  const float* ak  = (const float*)d_in[8];
  const float* gk  = (const float*)d_in[9];
  const float* bek = (const float*)d_in[10];
  const float* Wv  = (const float*)d_in[11];
  const float* bv  = (const float*)d_in[12];
  const float* av  = (const float*)d_in[13];
  const float* gv  = (const float*)d_in[14];
  const float* bev = (const float*)d_in[15];
  const float* Wp  = (const float*)d_in[16];
  const float* bp  = (const float*)d_in[17];
  const float* ap  = (const float*)d_in[18];
  const float* gp  = (const float*)d_in[19];
  const float* bep = (const float*)d_in[20];
  float* out = (float*)d_out;

  _Float16* qh = (_Float16*)d_ws;
  _Float16* kh = qh + (size_t)16 * 1024 * 512;
  _Float16* vh = kh + (size_t)16 * 1024 * 512;
  _Float16* S  = vh + (size_t)16 * 1024 * 2048;
  _Float16* oh = S  + (size_t)16 * 1024 * 1024;

  qkv_kernel<<<4096, 256, 0, stream>>>(x, Wq, bq, aq, gq, beq, Wk, bk, ak, gk,
                                       bek, Wv, bv, av, gv, bev, qh, kh, vh);
  scores_kernel<<<dim3(8, 8, 16), 256, 0, stream>>>(qh, kh, S);
  softmax_kernel<<<dim3(1024, 16), 256, 0, stream>>>(S);
  pv_kernel<<<dim3(16, 8, 16), 256, 0, stream>>>(S, vh, oh);
  proj_kernel<<<4096, 256, 0, stream>>>(oh, x, Wp, bp, ap, gp, bep, out);
}
